// MoeLayer_40888088658468
// MI455X (gfx1250) — matrix operand order repaired and checked by reading, not compiled, not run on hardware
//
#include <hip/hip_runtime.h>
#include <hip/hip_bf16.h>

// MoE top-2, T=8192 D=1024 F=4096 E=8.
// Sparse dispatch (gather per expert) -> 4.0e11 FLOP vs 1.6e12 dense.
// bf16 WMMA (v_wmma_f32_16x16x32_bf16); native fp32<->bf16 converts.
// Register double-buffered global->LDS pipeline; B transpose via 2x ds_store_b128.

#define TTOK 8192
#define DDIM 1024
#define FDIM 4096
#define NEXP 8

typedef __attribute__((ext_vector_type(16))) __bf16     v16bf;
typedef __attribute__((ext_vector_type(8)))  float      v8f;
typedef __attribute__((ext_vector_type(4)))  float      f32x4;
typedef __attribute__((ext_vector_type(2)))  float      f32x2;
typedef __attribute__((ext_vector_type(4)))  unsigned   u32x4;

__device__ __forceinline__ u32x4 pack8(f32x4 lo, f32x4 hi) {   // 8x fp32 -> 8x bf16
    union { __bf16 h[8]; u32x4 q; } u;
    u.h[0] = (__bf16)lo[0]; u.h[1] = (__bf16)lo[1]; u.h[2] = (__bf16)lo[2]; u.h[3] = (__bf16)lo[3];
    u.h[4] = (__bf16)hi[0]; u.h[5] = (__bf16)hi[1]; u.h[6] = (__bf16)hi[2]; u.h[7] = (__bf16)hi[3];
    return u.q;
}

// ---- A fragment (16x32 bf16): lane<16 row=lane K=0..15;
//      lane>=16 row=lane-16 K=16..31. LDS tile [m][k], k-stride 32.
__device__ __forceinline__ v16bf ldsA_frag(const __bf16* lA, int row0) {
    const int lane = threadIdx.x & 31;
    const __bf16* p = lA + (row0 + (lane & 15)) * 32 + ((lane < 16) ? 0 : 16);
    union { u32x4 q[2]; v16bf v; } u;
    u.q[0] = *(const u32x4*)p;
    u.q[1] = *(const u32x4*)(p + 8);
    return u.v;
}
// ---- B fragment (32x16 bf16): lane<16 col=lane K=0..15; lane>=16 col=lane-16 K=16..31.
//      LDS tile stored transposed [n][k] -> 16 contiguous bf16 per lane.
__device__ __forceinline__ v16bf ldsB_frag(const __bf16* lB, int col0) {
    const int lane = threadIdx.x & 31;
    const __bf16* p = lB + (col0 + (lane & 15)) * 32 + ((lane < 16) ? 0 : 16);
    union { u32x4 q[2]; v16bf v; } u;
    u.q[0] = *(const u32x4*)p;
    u.q[1] = *(const u32x4*)(p + 8);
    return u.v;
}

// =============================== gating ===================================
__global__ void moe_gate_kernel(const float* __restrict__ x,
                                const float* __restrict__ gw,
                                int* __restrict__ cursor,
                                int* __restrict__ tok,
                                float* __restrict__ wts) {
    const int wave = threadIdx.x >> 5;
    const int lane = threadIdx.x & 31;
    const int t = blockIdx.x * 8 + wave;
    float acc[NEXP];
#pragma unroll
    for (int e = 0; e < NEXP; ++e) acc[e] = 0.f;
    const float* xr = x + (size_t)t * DDIM;
    for (int d = lane; d < DDIM; d += 32) {
        const float xv = xr[d];
        const f32x4 g0 = *(const f32x4*)(gw + d * NEXP);
        const f32x4 g1 = *(const f32x4*)(gw + d * NEXP + 4);
        acc[0] = fmaf(xv, g0[0], acc[0]); acc[1] = fmaf(xv, g0[1], acc[1]);
        acc[2] = fmaf(xv, g0[2], acc[2]); acc[3] = fmaf(xv, g0[3], acc[3]);
        acc[4] = fmaf(xv, g1[0], acc[4]); acc[5] = fmaf(xv, g1[1], acc[5]);
        acc[6] = fmaf(xv, g1[2], acc[6]); acc[7] = fmaf(xv, g1[3], acc[7]);
    }
#pragma unroll
    for (int off = 16; off > 0; off >>= 1) {
#pragma unroll
        for (int e = 0; e < NEXP; ++e) acc[e] += __shfl_xor(acc[e], off, 32);
    }
    if (lane == 0) {
        int e0 = 0; float l0 = acc[0];
#pragma unroll
        for (int e = 1; e < NEXP; ++e) if (acc[e] > l0) { l0 = acc[e]; e0 = e; }
        int e1 = (e0 == 0) ? 1 : 0; float l1 = acc[e1];
#pragma unroll
        for (int e = 0; e < NEXP; ++e) if (e != e0 && acc[e] > l1) { l1 = acc[e]; e1 = e; }
        const float w0 = 1.f / (1.f + __expf(l1 - l0));  // softmax over top-2
        const float w1 = 1.f - w0;
        const int p0 = atomicAdd(cursor + e0, 1);
        tok[e0 * TTOK + p0] = t; wts[e0 * TTOK + p0] = w0;
        const int p1 = atomicAdd(cursor + e1, 1);
        tok[e1 * TTOK + p1] = t; wts[e1 * TTOK + p1] = w1;
    }
}

// ====================== up GEMM: H = bf16( Xg @ W ) =======================
// A: gathered tokens [rows, D] fp32 -> bf16; B: W [D, F] fp32 -> bf16.
__global__ void __launch_bounds__(256)
moe_gemm_up(const float* __restrict__ x, const float* __restrict__ W,
            const int* __restrict__ tok, const int* __restrict__ cntp,
            __bf16* __restrict__ H) {
    const int cnt = cntp[0];
    if ((int)(blockIdx.y * 128) >= ((cnt + 127) & ~127)) return;

    __shared__ alignas(16) __bf16 lA[128 * 32];
    __shared__ alignas(16) __bf16 lB[128 * 32];

    const int tid = threadIdx.x;
    // A stage: 2 threads/row, 16 contiguous k each
    const int arow = tid >> 1, akc = (tid & 1) << 4;
    const int token = tok[blockIdx.y * 128 + arow];
    const float* ga = x + (size_t)token * DDIM + akc;
    // B stage: thread owns 2 n-cols x 8 k-rows (coalesced f32x2 loads,
    // packed transpose -> two ds_store_b128 per tile)
    const int bn2 = (tid & 63) << 1, bkq = (tid >> 6) << 3;
    const float* gb = W + (size_t)bkq * FDIM + blockIdx.x * 128 + bn2;

    const int wave = tid >> 5, lane = tid & 31;
    const int wm = wave >> 1, wn = wave & 1;                       // 4x2 wave grid

    v8f acc[2][4] = {};

    f32x4 ra0, ra1, ra2, ra3;
    float rb0[8], rb1[8];
#define UP_LOAD(K0)                                                        \
    do {                                                                   \
        ra0 = *(const f32x4*)(ga + (K0) + 0);                              \
        ra1 = *(const f32x4*)(ga + (K0) + 4);                              \
        ra2 = *(const f32x4*)(ga + (K0) + 8);                              \
        ra3 = *(const f32x4*)(ga + (K0) + 12);                             \
        _Pragma("unroll")                                                  \
        for (int kk = 0; kk < 8; ++kk) {                                   \
            const f32x2 v = *(const f32x2*)(gb + (size_t)((K0) + kk) * FDIM); \
            rb0[kk] = v[0]; rb1[kk] = v[1];                                \
        }                                                                  \
    } while (0)

    UP_LOAD(0);
    for (int k0 = 0; k0 < DDIM; k0 += 32) {
        __syncthreads();                       // previous compute done with LDS
        *(u32x4*)&lA[arow * 32 + akc]     = pack8(ra0, ra1);
        *(u32x4*)&lA[arow * 32 + akc + 8] = pack8(ra2, ra3);
        {
            union { __bf16 h[8]; u32x4 q; } p0, p1;
#pragma unroll
            for (int kk = 0; kk < 8; ++kk) { p0.h[kk] = (__bf16)rb0[kk]; p1.h[kk] = (__bf16)rb1[kk]; }
            *(u32x4*)&lB[(bn2 + 0) * 32 + bkq] = p0.q;
            *(u32x4*)&lB[(bn2 + 1) * 32 + bkq] = p1.q;
        }
        __syncthreads();                       // tile ready
        if (k0 + 32 < DDIM) UP_LOAD(k0 + 32);  // next tile flies under WMMA below
        if (k0 + 64 < DDIM) {                  // global_prefetch_b8, near scope
            __builtin_prefetch(ga + k0 + 64, 0, 3);
            __builtin_prefetch(gb + (size_t)(k0 + 64) * FDIM, 0, 3);
        }

        const v16bf fa0 = ldsA_frag(lA, wm * 32);
        const v16bf fa1 = ldsA_frag(lA, wm * 32 + 16);
#pragma unroll
        for (int j = 0; j < 4; ++j) {
            const v16bf fb = ldsB_frag(lB, wn * 64 + j * 16);
            acc[0][j] = __builtin_amdgcn_wmma_f32_16x16x32_bf16(false, fa0, false, fb, (short)0, acc[0][j], false, false);
            acc[1][j] = __builtin_amdgcn_wmma_f32_16x16x32_bf16(false, fa1, false, fb, (short)0, acc[1][j], false, false);
        }
    }
#undef UP_LOAD

    // C layout (ISA 7.12.2): VGPR r -> M = r (lanes 0-15) or r+8 (lanes 16-31), N = lane&15
    const int roff = (lane < 16) ? 0 : 8;
    const int ncol = lane & 15;
#pragma unroll
    for (int i = 0; i < 2; ++i) {
        const int mb = blockIdx.y * 128 + wm * 32 + i * 16 + roff;
#pragma unroll
        for (int j = 0; j < 4; ++j) {
            const int nb = blockIdx.x * 128 + wn * 64 + j * 16 + ncol;
#pragma unroll
            for (int r = 0; r < 8; ++r)
                H[(size_t)(mb + r) * FDIM + nb] = (__bf16)acc[i][j][r];
        }
    }
}

// ============ down GEMM: out[tok] += wt * ( silu(H1)*H3 @ W2 ) ============
__global__ void __launch_bounds__(256)
moe_gemm_down(const __bf16* __restrict__ H1,
              const __bf16* __restrict__ H3,
              const float* __restrict__ W2,
              const int* __restrict__ tok, const float* __restrict__ wts,
              const int* __restrict__ cntp,
              float* __restrict__ out) {
    const int cnt = cntp[0];
    if ((int)(blockIdx.y * 128) >= ((cnt + 127) & ~127)) return;

    __shared__ alignas(16) __bf16 lA[128 * 32];
    __shared__ alignas(16) __bf16 lB[128 * 32];

    const int tid = threadIdx.x;
    const int arow = tid >> 1, akc = (tid & 1) << 4;
    const int slot0 = blockIdx.y * 128 + arow;
    const __bf16* g1 = H1 + (size_t)slot0 * FDIM + akc;
    const __bf16* g3 = H3 + (size_t)slot0 * FDIM + akc;
    const int bn2 = (tid & 63) << 1, bkq = (tid >> 6) << 3;
    const float* gb = W2 + (size_t)bkq * DDIM + blockIdx.x * 128 + bn2;

    const int wave = tid >> 5, lane = tid & 31;
    const int wm = wave >> 1, wn = wave & 1;

    v8f acc[2][4] = {};

    union { u32x4 q[2]; __bf16 s[16]; } r1, r3;
    float rb0[8], rb1[8];
#define DN_LOAD(K0)                                                        \
    do {                                                                   \
        r1.q[0] = *(const u32x4*)(g1 + (K0));                              \
        r1.q[1] = *(const u32x4*)(g1 + (K0) + 8);                          \
        r3.q[0] = *(const u32x4*)(g3 + (K0));                              \
        r3.q[1] = *(const u32x4*)(g3 + (K0) + 8);                          \
        _Pragma("unroll")                                                  \
        for (int kk = 0; kk < 8; ++kk) {                                   \
            const f32x2 v = *(const f32x2*)(gb + (size_t)((K0) + kk) * DDIM); \
            rb0[kk] = v[0]; rb1[kk] = v[1];                                \
        }                                                                  \
    } while (0)

    DN_LOAD(0);
    for (int k0 = 0; k0 < FDIM; k0 += 32) {
        __syncthreads();
        {
            union { __bf16 h[16]; u32x4 q[2]; } ph;
#pragma unroll
            for (int j = 0; j < 16; ++j) {         // fused SwiGLU on the A tile
                const float a = (float)r1.s[j];
                const float b = (float)r3.s[j];
                ph.h[j] = (__bf16)((a / (1.f + __expf(-a))) * b);
            }
            *(u32x4*)&lA[arow * 32 + akc]     = ph.q[0];
            *(u32x4*)&lA[arow * 32 + akc + 8] = ph.q[1];
        }
        {
            union { __bf16 h[8]; u32x4 q; } p0, p1;
#pragma unroll
            for (int kk = 0; kk < 8; ++kk) { p0.h[kk] = (__bf16)rb0[kk]; p1.h[kk] = (__bf16)rb1[kk]; }
            *(u32x4*)&lB[(bn2 + 0) * 32 + bkq] = p0.q;
            *(u32x4*)&lB[(bn2 + 1) * 32 + bkq] = p1.q;
        }
        __syncthreads();
        if (k0 + 32 < FDIM) DN_LOAD(k0 + 32);
        if (k0 + 64 < FDIM) {
            __builtin_prefetch(g1 + k0 + 64, 0, 3);
            __builtin_prefetch(g3 + k0 + 64, 0, 3);
            __builtin_prefetch(gb + (size_t)(k0 + 64) * DDIM, 0, 3);
        }

        const v16bf fa0 = ldsA_frag(lA, wm * 32);
        const v16bf fa1 = ldsA_frag(lA, wm * 32 + 16);
#pragma unroll
        for (int j = 0; j < 4; ++j) {
            const v16bf fb = ldsB_frag(lB, wn * 64 + j * 16);
            acc[0][j] = __builtin_amdgcn_wmma_f32_16x16x32_bf16(false, fa0, false, fb, (short)0, acc[0][j], false, false);
            acc[1][j] = __builtin_amdgcn_wmma_f32_16x16x32_bf16(false, fa1, false, fb, (short)0, acc[1][j], false, false);
        }
    }
#undef DN_LOAD

    const int roff = (lane < 16) ? 0 : 8;
    const int ncol = lane & 15;
#pragma unroll
    for (int i = 0; i < 2; ++i) {
        const int mb = blockIdx.y * 128 + wm * 32 + i * 16 + roff;
#pragma unroll
        for (int j = 0; j < 4; ++j) {
            const int nb = blockIdx.x * 128 + wn * 64 + j * 16 + ncol;
#pragma unroll
            for (int r = 0; r < 8; ++r) {
                const int slot = mb + r;
                if (slot < cnt) {                   // padding slots never touch out
                    const int t = tok[slot];
                    const float w = wts[slot];
                    float* o = out + (size_t)t * DDIM + nb;
                    *o += w * acc[i][j][r];         // single writer per (t,n) per launch
                }
            }
        }
    }
}

// ================================ host ====================================
extern "C" void kernel_launch(void* const* d_in, const int* in_sizes, int n_in,
                              void* d_out, int out_size, void* d_ws, size_t ws_size,
                              hipStream_t stream) {
    (void)in_sizes; (void)n_in; (void)out_size; (void)ws_size;
    const float* x  = (const float*)d_in[0];   // [T, D]
    const float* gw = (const float*)d_in[1];   // [D, E]
    const float* w1 = (const float*)d_in[2];   // [E, D, F]
    const float* w3 = (const float*)d_in[3];   // [E, D, F]
    const float* w2 = (const float*)d_in[4];   // [E, F, D]

    char* ws = (char*)d_ws;
    int*   cursor = (int*)ws;                                          // NEXP counters
    int*   tok    = (int*)(ws + 1024);                                 // [E][T]
    float* wts    = (float*)(ws + 1024 + sizeof(int) * NEXP * TTOK);   // [E][T]
    const size_t hoff = (size_t)1 << 20;
    __bf16* H1 = (__bf16*)(ws + hoff);                                 // 64 MB bf16
    __bf16* H3 = (__bf16*)(ws + hoff + (size_t)TTOK * FDIM * 2);

    // zero counters + lists (token 0 / weight 0 sentinels for padding) and out
    hipMemsetAsync(d_ws, 0, 1024 + (size_t)8 * NEXP * TTOK, stream);
    hipMemsetAsync(d_out, 0, (size_t)TTOK * DDIM * sizeof(float), stream);

    moe_gate_kernel<<<TTOK / 8, 256, 0, stream>>>(x, gw, cursor, tok, wts);

    dim3 gu(FDIM / 128, TTOK / 128);   // 32 x 64, early-exit past padded count
    dim3 gd(DDIM / 128, TTOK / 128);   // 8 x 64
    for (int e = 0; e < NEXP; ++e) {   // serialized -> deterministic out accumulation
        const float* W1e = w1 + (size_t)e * DDIM * FDIM;
        const float* W3e = w3 + (size_t)e * DDIM * FDIM;
        const float* W2e = w2 + (size_t)e * FDIM * DDIM;
        moe_gemm_up<<<gu, 256, 0, stream>>>(x, W1e, tok + e * TTOK, cursor + e, H1);
        moe_gemm_up<<<gu, 256, 0, stream>>>(x, W3e, tok + e * TTOK, cursor + e, H3);
        moe_gemm_down<<<gd, 256, 0, stream>>>(H1, H3, W2e, tok + e * TTOK, wts + e * TTOK,
                                              cursor + e, (float*)d_out);
    }
}
